// StaticHDC_42159398978225
// MI455X (gfx1250) — 8-bit matrix operand order repaired and checked by reading, compiled, not run on hardware
//
#include <hip/hip_runtime.h>

#define D_TOT  10000
#define NPIX   784
#define NV     10
#define NB     32
#define KT8    7
#define TT     20
#define NCHUNK (D_TOT / 16)

typedef __attribute__((ext_vector_type(16))) int   v16i;
typedef __attribute__((ext_vector_type(8)))  float v8f;

#define FP8_ONE     0x38u
#define FP8_NEGONE  0xB8u

#define AFRAG_BYTES ((size_t)TT * KT8 * 32 * 64)
#define SIMS_FLOATS (NB * NV)

__global__ void hdc_zero(float* __restrict__ sims) {
    int i = blockIdx.x * blockDim.x + threadIdx.x;
    if (i < SIMS_FLOATS) sims[i] = 0.0f;
}

__global__ void hdc_build_afrag(const float* __restrict__ x,
                                unsigned int* __restrict__ afrag) {
    int gid = blockIdx.x * blockDim.x + threadIdx.x;
    if (gid >= TT * KT8 * 32) return;
    int lane = gid & 31;
    int kt   = (gid >> 5) % KT8;
    int t    = (gid >> 5) / KT8;
    int v    = t >> 1;
    int b    = (t & 1) * 16 + (lane & 15);
    int lhs  = (lane >> 4) ? 16 : 0;

    v16i r;
    for (int w = 0; w < 16; ++w) {
        int half = w >> 3, wp = w & 7;
        int kbase = kt * 128 + 64 * half + ((wp & 4) << 3) + ((wp & 2) << 2)
                  + ((wp & 1) << 2) + lhs;
        unsigned u = 0u;
        for (int c = 0; c < 4; ++c) {
            int p = kbase + c;
            if (p < NPIX) {
                float xv = x[(size_t)b * NPIX + p];
                int idx = (int)(xv * 10.0f);
                idx = idx < 0 ? 0 : (idx > NV - 1 ? NV - 1 : idx);
                if (idx == v) u |= FP8_ONE << (8 * c);
            }
        }
        r[w] = (int)u;
    }
    *(v16i*)(afrag + (size_t)gid * 16) = r;
}

__global__ void __launch_bounds__(32)
hdc_main(const float* __restrict__ pos,
         const float* __restrict__ vw,
         const float* __restrict__ clf,
         const int*   __restrict__ afrag,
         float* __restrict__ sims) {
    __shared__ __attribute__((aligned(16))) float enc_lds[NB * 16];

    const int lane  = threadIdx.x;
    const int chunk = blockIdx.x;
    if (chunk >= NCHUNK) return;
    const int dbase = chunk * 16;
    const int nlo   = lane & 15;
    const int ksub  = lane >> 4;
    const int dcol  = dbase + nlo;

    v16i bfrag[KT8];
    for (int kt = 0; kt < KT8; ++kt) {
        for (int g = 0; g < 4; ++g) {
            int pbase = kt * 128 + g * 32 + ksub * 16;
            unsigned u[4] = {0u, 0u, 0u, 0u};
            if (pbase < NPIX) {
                const float4* src =
                    (const float4*)(pos + (size_t)dcol * NPIX + pbase);
                for (int q = 0; q < 4; ++q) {
                    float4 f = src[q];
                    unsigned b0 = f.x < 0.f ? FP8_NEGONE : FP8_ONE;
                    unsigned b1 = f.y < 0.f ? FP8_NEGONE : FP8_ONE;
                    unsigned b2 = f.z < 0.f ? FP8_NEGONE : FP8_ONE;
                    unsigned b3 = f.w < 0.f ? FP8_NEGONE : FP8_ONE;
                    u[q] = b0 | (b1 << 8) | (b2 << 16) | (b3 << 24);
                }
            }
            bfrag[kt][g * 4 + 0] = (int)u[0];
            bfrag[kt][g * 4 + 1] = (int)u[1];
            bfrag[kt][g * 4 + 2] = (int)u[2];
            bfrag[kt][g * 4 + 3] = (int)u[3];
        }
    }

    v8f accLo = {};
    v8f accHi = {};
    for (int v = 0; v < NV; ++v) {
        v8f c0 = {};
        v8f c1 = {};
        const v16i* a0 =
            (const v16i*)afrag + ((size_t)(2 * v) * KT8 * 32 + lane);
        const v16i* a1 = a0 + (size_t)KT8 * 32;
        for (int kt = 0; kt < KT8; ++kt) {
            v16i af0 = a0[kt * 32];
            v16i af1 = a1[kt * 32];
            c0 = __builtin_amdgcn_wmma_f32_16x16x128_fp8_fp8(
                     af0, bfrag[kt], (short)0, c0, false, false);
            c1 = __builtin_amdgcn_wmma_f32_16x16x128_fp8_fp8(
                     af1, bfrag[kt], (short)0, c1, false, false);
        }
        float wv = vw[(size_t)v * D_TOT + dcol];
        for (int j = 0; j < 8; ++j) {
            accLo[j] += wv * c0[j];
            accHi[j] += wv * c1[j];
        }
    }

    const int ms = ksub * 8;
    for (int j = 0; j < 8; ++j) {
        float s = accLo[j];
        enc_lds[(j + ms) * 16 + nlo]      = s > 0.f ? 1.f : (s < 0.f ? -1.f : 0.f);
        s = accHi[j];
        enc_lds[(16 + j + ms) * 16 + nlo] = s > 0.f ? 1.f : (s < 0.f ? -1.f : 0.f);
    }
    __syncthreads();

    float sim[NV];
    const float* er = enc_lds + lane * 16;
    for (int c = 0; c < NV; ++c) {
        const float* cr = clf + (size_t)c * D_TOT + dbase;
        float a = 0.f;
        for (int dd = 0; dd < 16; ++dd) a += er[dd] * cr[dd];
        sim[c] = a;
    }
    for (int c = 0; c < NV; ++c) atomicAdd(&sims[lane * NV + c], sim[c]);
}

__global__ void hdc_argmax(const float* __restrict__ sims, float* __restrict__ out) {
    int b = threadIdx.x;
    if (b >= NB) return;
    float best = sims[b * NV];
    int bi = 0;
    for (int c = 1; c < NV; ++c) {
        float s = sims[b * NV + c];
        if (s > best) { best = s; bi = c; }
    }
    out[b] = (float)bi;
}

extern "C" void kernel_launch(void* const* d_in, const int* in_sizes, int n_in,
                              void* d_out, int out_size, void* d_ws, size_t ws_size,
                              hipStream_t stream) {
    (void)in_sizes; (void)n_in; (void)out_size; (void)ws_size;
    const float* x   = (const float*)d_in[0];
    const float* vw  = (const float*)d_in[1];
    const float* pos = (const float*)d_in[2];
    const float* clf = (const float*)d_in[3];

    unsigned int* afrag = (unsigned int*)d_ws;
    float*        sims  = (float*)((char*)d_ws + AFRAG_BYTES);

    hdc_zero<<<1, 320, 0, stream>>>(sims);
    hdc_build_afrag<<<(TT * KT8 * 32 + 255) / 256, 256, 0, stream>>>(x, afrag);
    hdc_main<<<NCHUNK, 32, 0, stream>>>(pos, vw, clf, (const int*)afrag, sims);
    hdc_argmax<<<1, 32, 0, stream>>>(sims, (float*)d_out);
}
